// EvoSABlock_65223373357209
// MI455X (gfx1250) — matrix operand order repaired and checked by reading, not compiled, not run on hardware
//
#include <hip/hip_runtime.h>
#include <hip/hip_bf16.h>
#include <math.h>

// ---------------- problem constants ----------------
#define B_    4
#define C_    384
#define T_    8
#define H_    28
#define W_    28
#define N_TOK 6272            // T*H*W
#define NH    6
#define HD    64
#define MLP   1536
#define NKEEP 3136            // N * PRUNE_RATIO
#define M_TOK 3138            // NKEEP + 2 (cls + rep)
#define MP    3200            // padded row count (multiple of 64)
#define KEY_END 3168          // multiple of 32 covering M_TOK
#define QT_COUNT 197          // ceil(M_TOK / 16)

typedef __attribute__((ext_vector_type(16))) _Float16 v16h;
typedef __attribute__((ext_vector_type(8)))  _Float16 v8h;
typedef __attribute__((ext_vector_type(8)))  float    v8f;

// ---------------- WMMA helpers ----------------
// A fragment (16x32 f16): lane L holds row M=L%16; halves 0..15 = K in
// [2*l16, 2*l16+16), l16 = (L>=16)*8.
__device__ __forceinline__ v16h load_a_frag(const _Float16* row, int k0, int l16) {
  v8h lo = *(const v8h*)(row + k0 + (l16 << 1));
  v8h hi = *(const v8h*)(row + k0 + (l16 << 1) + 8);
  v16h a;
#pragma unroll
  for (int i = 0; i < 8; ++i) { a[i] = lo[i]; a[i + 8] = hi[i]; }
  return a;
}

__device__ __forceinline__ v8f wmma16(v16h a, v16h b, v8f c) {
  return __builtin_amdgcn_wmma_f32_16x16x32_f16(false, a, false, b, (short)0, c, false, false);
}

// Register-blocked, software-pipelined tile: one A fragment (16 rows) x
// 4 B tiles (64 cols); next K-step's fragments prefetched while the
// current step's 4 WMMAs execute.
// A: row-major f16, stride lda.  Wt: (N x K) row-major f16, stride ldw
// (B fragment: lane L holds col N=L%16, 16 contiguous K halves at (L>=16)*16).
__device__ __forceinline__ void gemm_tile4(const _Float16* __restrict__ A, int lda,
                                           const _Float16* __restrict__ Wt, int ldw,
                                           int m0, int n0, int K, int lane,
                                           v8f acc[4]) {
  int l16  = (lane >> 4) << 3;
  int koff = (lane >> 4) << 4;
  const _Float16* arow = A  + (size_t)(m0 + (lane & 15)) * lda;
  const _Float16* w0 = Wt + (size_t)(n0 +  0 + (lane & 15)) * ldw + koff;
  const _Float16* w1 = Wt + (size_t)(n0 + 16 + (lane & 15)) * ldw + koff;
  const _Float16* w2 = Wt + (size_t)(n0 + 32 + (lane & 15)) * ldw + koff;
  const _Float16* w3 = Wt + (size_t)(n0 + 48 + (lane & 15)) * ldw + koff;
  v16h a  = load_a_frag(arow, 0, l16);
  v16h b0 = *(const v16h*)(w0);
  v16h b1 = *(const v16h*)(w1);
  v16h b2 = *(const v16h*)(w2);
  v16h b3 = *(const v16h*)(w3);
  int steps = K >> 5;
  for (int s = 0; s < steps - 1; ++s) {
    int kn = (s + 1) << 5;
    v16h an = load_a_frag(arow, kn, l16);
    v16h c0 = *(const v16h*)(w0 + kn);
    v16h c1 = *(const v16h*)(w1 + kn);
    v16h c2 = *(const v16h*)(w2 + kn);
    v16h c3 = *(const v16h*)(w3 + kn);
    acc[0] = wmma16(a, b0, acc[0]);
    acc[1] = wmma16(a, b1, acc[1]);
    acc[2] = wmma16(a, b2, acc[2]);
    acc[3] = wmma16(a, b3, acc[3]);
    a = an; b0 = c0; b1 = c1; b2 = c2; b3 = c3;
  }
  acc[0] = wmma16(a, b0, acc[0]);
  acc[1] = wmma16(a, b1, acc[1]);
  acc[2] = wmma16(a, b2, acc[2]);
  acc[3] = wmma16(a, b3, acc[3]);
}

// ---------------- elementwise / prep kernels ----------------
__global__ void k_f2h(const float* __restrict__ s, _Float16* __restrict__ d, int n) {
  int i = blockIdx.x * 256 + threadIdx.x;
  if (i < n) d[i] = (_Float16)s[i];
}

// depthwise 3x3x3 conv (pad 1) + residual; emit tokens xt[b][n][c]
__global__ void k_conv(const float* __restrict__ x, const float* __restrict__ cw,
                       const float* __restrict__ cb, float* __restrict__ xt) {
  int idx = blockIdx.x * 256 + threadIdx.x;
  if (idx >= B_ * C_ * N_TOK) return;
  int w = idx % W_;  int r0 = idx / W_;
  int h = r0 % H_;   int r1 = r0 / H_;
  int t = r1 % T_;   int r2 = r1 / T_;
  int c = r2 % C_;   int b = r2 / C_;
  const float* xb = x + ((size_t)b * C_ + c) * (T_ * H_ * W_);
  const float* wc = cw + c * 27;
  float s = cb[c];
  for (int dt = -1; dt <= 1; ++dt) {
    int tt = t + dt; if (tt < 0 || tt >= T_) continue;
    for (int dh = -1; dh <= 1; ++dh) {
      int hh = h + dh; if (hh < 0 || hh >= H_) continue;
      for (int dw = -1; dw <= 1; ++dw) {
        int ww = w + dw; if (ww < 0 || ww >= W_) continue;
        s += xb[(tt * H_ + hh) * W_ + ww] * wc[(dt + 1) * 9 + (dh + 1) * 3 + (dw + 1)];
      }
    }
  }
  int n = (t * H_ + h) * W_ + w;
  xt[((size_t)b * N_TOK + n) * C_ + c] = xb[(t * H_ + h) * W_ + w] + s;
}

// stable descending rank == position produced by argsort(-ga)
__global__ void k_rank(const float* __restrict__ ga, int* __restrict__ rank) {
  int idx = blockIdx.x * 256 + threadIdx.x;
  if (idx >= B_ * N_TOK) return;
  int b = idx / N_TOK, i = idx % N_TOK;
  const float* g = ga + (size_t)b * N_TOK;
  float gi = g[i];
  int r = 0;
  for (int j = 0; j < N_TOK; ++j) {
    float gj = g[j];
    r += (gj > gi) || (gj == gi && j < i);
  }
  rank[idx] = r;
}

__global__ void k_sdrop(const float* __restrict__ ga, const int* __restrict__ rank,
                        float* __restrict__ sdrop) {
  __shared__ float red[256];
  int b = blockIdx.x;
  float s = 0.f;
  for (int i = threadIdx.x; i < N_TOK; i += 256)
    if (rank[b * N_TOK + i] >= NKEEP) s += ga[b * N_TOK + i];
  red[threadIdx.x] = s; __syncthreads();
  for (int o = 128; o > 0; o >>= 1) {
    if (threadIdx.x < o) red[threadIdx.x] += red[threadIdx.x + o];
    __syncthreads();
  }
  if (threadIdx.x == 0) sdrop[b] = red[0];
}

__global__ void k_build_cls(const float* __restrict__ cls, float* __restrict__ xa) {
  int idx = blockIdx.x * 256 + threadIdx.x;
  if (idx >= B_ * C_) return;
  int b = idx / C_, c = idx % C_;
  xa[((size_t)b * MP) * C_ + c] = cls[b * C_ + c];
}

__global__ void k_build_xa(const float* __restrict__ xt, const int* __restrict__ rank,
                           float* __restrict__ xa) {
  int idx = blockIdx.x * 256 + threadIdx.x;
  if (idx >= B_ * N_TOK * C_) return;
  int c = idx % C_;  int t = idx / C_;
  int i = t % N_TOK; int b = t / N_TOK;
  int r = rank[b * N_TOK + i];
  if (r < NKEEP)
    xa[((size_t)b * MP + 1 + r) * C_ + c] = xt[((size_t)b * N_TOK + i) * C_ + c];
}

// rep token: weighted sum of dropped tokens, one block per (b, c)
__global__ void k_rep(const float* __restrict__ xt, const float* __restrict__ ga,
                      const int* __restrict__ rank, const float* __restrict__ sdrop,
                      float* __restrict__ xa) {
  __shared__ float red[256];
  int b = blockIdx.x / C_, c = blockIdx.x % C_;
  float s = 0.f;
  for (int i = threadIdx.x; i < N_TOK; i += 256)
    if (rank[b * N_TOK + i] >= NKEEP)
      s += ga[b * N_TOK + i] * xt[((size_t)b * N_TOK + i) * C_ + c];
  red[threadIdx.x] = s; __syncthreads();
  for (int o = 128; o > 0; o >>= 1) {
    if (threadIdx.x < o) red[threadIdx.x] += red[threadIdx.x + o];
    __syncthreads();
  }
  if (threadIdx.x == 0)
    xa[((size_t)b * MP + (M_TOK - 1)) * C_ + c] = red[0] / sdrop[b];
}

// LayerNorm row -> f16; zero pad rows
__global__ void k_ln(const float* __restrict__ xa, const float* __restrict__ g,
                     const float* __restrict__ be, _Float16* __restrict__ h16) {
  __shared__ float red[128];
  int row = blockIdx.x;
  int m = row % MP;
  size_t off = (size_t)row * C_;
  if (m >= M_TOK) {            // uniform per block
    for (int c = threadIdx.x; c < C_; c += 128) h16[off + c] = (_Float16)0.f;
    return;
  }
  float s = 0.f;
  for (int c = threadIdx.x; c < C_; c += 128) s += xa[off + c];
  red[threadIdx.x] = s; __syncthreads();
  for (int o = 64; o > 0; o >>= 1) {
    if (threadIdx.x < o) red[threadIdx.x] += red[threadIdx.x + o];
    __syncthreads();
  }
  float mean = red[0] / (float)C_;
  __syncthreads();
  float v = 0.f;
  for (int c = threadIdx.x; c < C_; c += 128) { float d = xa[off + c] - mean; v += d * d; }
  red[threadIdx.x] = v; __syncthreads();
  for (int o = 64; o > 0; o >>= 1) {
    if (threadIdx.x < o) red[threadIdx.x] += red[threadIdx.x + o];
    __syncthreads();
  }
  float rs = rsqrtf(red[0] / (float)C_ + 1e-5f);
  for (int c = threadIdx.x; c < C_; c += 128)
    h16[off + c] = (_Float16)((xa[off + c] - mean) * rs * g[c] + be[c]);
}

// ---------------- GEMM kernels (WMMA, 16x64 per wave) ----------------
// QKV: N=1152, scatter to q16/k16 [bh][m][d] and vt16 [bh][d][m]
__global__ void k_gemm_qkv(const _Float16* __restrict__ A, const _Float16* __restrict__ Wt,
                           _Float16* __restrict__ q16, _Float16* __restrict__ k16,
                           _Float16* __restrict__ vt16) {
  int lane = threadIdx.x & 31, wave = threadIdx.x >> 5;
  int m0 = blockIdx.y * 64 + wave * 16;
  int n0 = blockIdx.x * 64;
  v8f acc[4] = {{}, {}, {}, {}};
  gemm_tile4(A, C_, Wt, C_, m0, n0, C_, lane, acc);
  int rbase = m0 + ((lane >> 4) << 3);
#pragma unroll
  for (int j = 0; j < 4; ++j) {
    int n = n0 + j * 16 + (lane & 15);
    int s = n / C_, rem = n % C_;
    int h = rem / HD, d = rem % HD;
#pragma unroll
    for (int r = 0; r < 8; ++r) {
      int row = rbase + r;
      int b = row / MP, m = row % MP;
      _Float16 val = (_Float16)acc[j][r];
      size_t bh = (size_t)(b * NH + h);
      if (s == 0)      q16[(bh * MP + m) * HD + d] = val;
      else if (s == 1) k16[(bh * MP + m) * HD + d] = val;
      else             vt16[(bh * HD + d) * MP + m] = val;
    }
  }
}

// generic bias GEMM (proj, fc2): f32 out
__global__ void k_gemm_bias(const _Float16* __restrict__ A, const _Float16* __restrict__ Wt,
                            const float* __restrict__ bias, float* __restrict__ out,
                            int K, int N) {
  int lane = threadIdx.x & 31, wave = threadIdx.x >> 5;
  int m0 = blockIdx.y * 64 + wave * 16;
  int n0 = blockIdx.x * 64;
  v8f acc[4] = {{}, {}, {}, {}};
  gemm_tile4(A, K, Wt, K, m0, n0, K, lane, acc);
  int rbase = m0 + ((lane >> 4) << 3);
#pragma unroll
  for (int j = 0; j < 4; ++j) {
    int n = n0 + j * 16 + (lane & 15);
    float bv = bias[n];
#pragma unroll
    for (int r = 0; r < 8; ++r)
      out[(size_t)(rbase + r) * N + n] = acc[j][r] + bv;
  }
}

// fc1: bias + exact GELU, f16 out
__global__ void k_gemm_fc1(const _Float16* __restrict__ A, const _Float16* __restrict__ Wt,
                           const float* __restrict__ bias, _Float16* __restrict__ out) {
  int lane = threadIdx.x & 31, wave = threadIdx.x >> 5;
  int m0 = blockIdx.y * 64 + wave * 16;
  int n0 = blockIdx.x * 64;
  v8f acc[4] = {{}, {}, {}, {}};
  gemm_tile4(A, C_, Wt, C_, m0, n0, C_, lane, acc);
  int rbase = m0 + ((lane >> 4) << 3);
#pragma unroll
  for (int j = 0; j < 4; ++j) {
    int n = n0 + j * 16 + (lane & 15);
    float bv = bias[n];
#pragma unroll
    for (int r = 0; r < 8; ++r) {
      float xg = acc[j][r] + bv;
      float gl = 0.5f * xg * (1.0f + erff(xg * 0.70710678118f));
      out[(size_t)(rbase + r) * MLP + n] = (_Float16)gl;
    }
  }
}

// ---------------- flash attention (WMMA) ----------------
__global__ void k_attn(const _Float16* __restrict__ q16, const _Float16* __restrict__ k16,
                       const _Float16* __restrict__ vt16, _Float16* __restrict__ o16) {
  __shared__ __align__(32) _Float16 plds[8 * 16 * 32];   // per-wave 16x32 P tile
  int lane = threadIdx.x & 31, wave = threadIdx.x >> 5;
  int b = blockIdx.y / NH, h = blockIdx.y % NH;
  int qt = blockIdx.x * 8 + wave;
  if (qt >= QT_COUNT) return;                            // no block barriers below
  int l16   = (lane >> 4) << 3;
  int koff  = (lane >> 4) << 4;
  int rbase = (lane >> 4) << 3;                          // D-layout row base
  size_t bh = (size_t)(b * NH + h);
  const _Float16* qrow = q16 + (bh * MP + (size_t)qt * 16 + (lane & 15)) * HD;
  v16h qf0 = load_a_frag(qrow, 0, l16);
  v16h qf1 = load_a_frag(qrow, 32, l16);
  const _Float16* kb = k16 + bh * MP * HD;
  const _Float16* vb = vt16 + bh * HD * MP;
  float rmax[8], rsum[8];
#pragma unroll
  for (int r = 0; r < 8; ++r) { rmax[r] = -1e30f; rsum[r] = 0.f; }
  v8f o0 = {}, o1 = {}, o2 = {}, o3 = {};
  _Float16* pl = plds + wave * 512;

  for (int kk = 0; kk < KEY_END; kk += 32) {
    v8f s0 = {}, s1 = {};
    {
      const _Float16* krow = kb + (size_t)(kk + (lane & 15)) * HD + koff;
      s0 = wmma16(qf0, *(const v16h*)krow, s0);
      s0 = wmma16(qf1, *(const v16h*)(krow + 32), s0);
    }
    {
      const _Float16* krow = kb + (size_t)(kk + 16 + (lane & 15)) * HD + koff;
      s1 = wmma16(qf0, *(const v16h*)krow, s1);
      s1 = wmma16(qf1, *(const v16h*)(krow + 32), s1);
    }
    bool ok0 = (kk + (lane & 15)) < M_TOK;
    bool ok1 = (kk + 16 + (lane & 15)) < M_TOK;
#pragma unroll
    for (int r = 0; r < 8; ++r) {
      float e0 = ok0 ? s0[r] * 0.125f : -1e30f;
      float e1 = ok1 ? s1[r] * 0.125f : -1e30f;
      float mloc = fmaxf(e0, e1);
#pragma unroll
      for (int off = 1; off < 16; off <<= 1) mloc = fmaxf(mloc, __shfl_xor(mloc, off));
      float nm = fmaxf(rmax[r], mloc);
      float fr = __expf(rmax[r] - nm);
      float p0 = __expf(e0 - nm), p1 = __expf(e1 - nm);
      float ps = p0 + p1;
#pragma unroll
      for (int off = 1; off < 16; off <<= 1) ps += __shfl_xor(ps, off);
      rsum[r] = rsum[r] * fr + ps;
      rmax[r] = nm;
      o0[r] *= fr; o1[r] *= fr; o2[r] *= fr; o3[r] *= fr;
      pl[(rbase + r) * 32 + (lane & 15)]      = (_Float16)p0;
      pl[(rbase + r) * 32 + 16 + (lane & 15)] = (_Float16)p1;
    }
    asm volatile("s_wait_dscnt 0" ::: "memory");         // order LDS write->read
    v16h pf;
    {
      v8h lo = *(const v8h*)(pl + (lane & 15) * 32 + (l16 << 1));
      v8h hi = *(const v8h*)(pl + (lane & 15) * 32 + (l16 << 1) + 8);
#pragma unroll
      for (int i = 0; i < 8; ++i) { pf[i] = lo[i]; pf[i + 8] = hi[i]; }
    }
    const _Float16* vrow = vb + (size_t)(lane & 15) * MP + kk + koff;
    o0 = wmma16(pf, *(const v16h*)(vrow),           o0);
    o1 = wmma16(pf, *(const v16h*)(vrow + 16 * MP), o1);
    o2 = wmma16(pf, *(const v16h*)(vrow + 32 * MP), o2);
    o3 = wmma16(pf, *(const v16h*)(vrow + 48 * MP), o3);
  }
  size_t orow0 = (size_t)b * MP + (size_t)qt * 16 + rbase;
#pragma unroll
  for (int r = 0; r < 8; ++r) {
    size_t base = (orow0 + r) * C_ + h * HD + (lane & 15);
    float inv = 1.0f / rsum[r];
    o16[base]      = (_Float16)(o0[r] * inv);
    o16[base + 16] = (_Float16)(o1[r] * inv);
    o16[base + 32] = (_Float16)(o2[r] * inv);
    o16[base + 48] = (_Float16)(o3[r] * inv);
  }
}

// cls-query softmax row (for ga update); one block per (b,h)
__global__ void k_cls_attn(const _Float16* __restrict__ q16, const _Float16* __restrict__ k16,
                           float* __restrict__ clsb) {
  __shared__ float logits[M_TOK];
  __shared__ float red[256];
  int b = blockIdx.x / NH, h = blockIdx.x % NH;
  size_t bh = (size_t)(b * NH + h);
  const _Float16* q0 = q16 + (bh * MP) * HD;
  const _Float16* kb = k16 + (bh * MP) * HD;
  for (int j = threadIdx.x; j < M_TOK; j += 256) {
    const _Float16* kr = kb + (size_t)j * HD;
    float dv = 0.f;
    for (int d = 0; d < HD; ++d) dv += (float)q0[d] * (float)kr[d];
    logits[j] = dv * 0.125f;
  }
  __syncthreads();
  float lm = -1e30f;
  for (int j = threadIdx.x; j < M_TOK; j += 256) lm = fmaxf(lm, logits[j]);
  red[threadIdx.x] = lm; __syncthreads();
  for (int o = 128; o > 0; o >>= 1) {
    if (threadIdx.x < o) red[threadIdx.x] = fmaxf(red[threadIdx.x], red[threadIdx.x + o]);
    __syncthreads();
  }
  float mx = red[0]; __syncthreads();
  float ls = 0.f;
  for (int j = threadIdx.x; j < M_TOK; j += 256) ls += __expf(logits[j] - mx);
  red[threadIdx.x] = ls; __syncthreads();
  for (int o = 128; o > 0; o >>= 1) {
    if (threadIdx.x < o) red[threadIdx.x] += red[threadIdx.x + o];
    __syncthreads();
  }
  float sum = red[0];
  for (int j = threadIdx.x; j < M_TOK; j += 256)
    if (j >= 1 && j <= NKEEP)
      atomicAdd(&clsb[b * NKEEP + (j - 1)], (__expf(logits[j] - mx) / sum) * (1.0f / NH));
}

// ---------------- epilogue kernels ----------------
__global__ void k_add(float* __restrict__ dst, const float* __restrict__ src, size_t n) {
  size_t i = (size_t)blockIdx.x * 256 + threadIdx.x;
  if (i < n) dst[i] += src[i];
}

__global__ void k_out_cls(const float* __restrict__ xa, float* __restrict__ out) {
  int idx = blockIdx.x * 256 + threadIdx.x;
  if (idx >= B_ * C_) return;
  int b = idx / C_, c = idx % C_;
  out[idx] = xa[((size_t)b * MP) * C_ + c];
}

__global__ void k_out_patch(const float* __restrict__ xa, const float* __restrict__ xt,
                            const int* __restrict__ rank, const float* __restrict__ obuf,
                            const float* __restrict__ m2, float* __restrict__ out) {
  int idx = blockIdx.x * 256 + threadIdx.x;
  if (idx >= B_ * C_ * N_TOK) return;
  int i = idx % N_TOK; int t = idx / N_TOK;
  int c = t % C_;      int b = t / C_;
  int r = rank[b * N_TOK + i];
  float v;
  if (r < NKEEP) {
    v = xa[((size_t)b * MP + 1 + r) * C_ + c];
  } else {
    size_t fu = ((size_t)b * MP + (M_TOK - 1)) * C_ + c;   // fast_update row
    v = xt[((size_t)b * N_TOK + i) * C_ + c] + obuf[fu] + m2[fu];
  }
  out[idx] = v;   // layout (B, C, N) == (B, C, T, H, W)
}

__global__ void k_out_ga(const float* __restrict__ ga, const int* __restrict__ rank,
                         const float* __restrict__ clsb, float* __restrict__ out) {
  int idx = blockIdx.x * 256 + threadIdx.x;
  if (idx >= B_ * N_TOK) return;
  int b = idx / N_TOK;
  int r = rank[idx];
  float g = ga[idx];
  out[idx] = (r < NKEEP) ? 0.5f * g + 0.5f * clsb[b * NKEEP + r] : g;
}

// ---------------- host ----------------
extern "C" void kernel_launch(void* const* d_in, const int* in_sizes, int n_in,
                              void* d_out, int out_size, void* d_ws, size_t ws_size,
                              hipStream_t stream) {
  (void)in_sizes; (void)n_in; (void)out_size; (void)ws_size;
  const float* cls    = (const float*)d_in[0];
  const float* x      = (const float*)d_in[1];
  const float* ga     = (const float*)d_in[2];
  const float* conv_w = (const float*)d_in[4];
  const float* conv_b = (const float*)d_in[5];
  const float* ln1g   = (const float*)d_in[6];
  const float* ln1b   = (const float*)d_in[7];
  const float* qkvw   = (const float*)d_in[8];
  const float* projw  = (const float*)d_in[9];
  const float* projb  = (const float*)d_in[10];
  const float* ln2g   = (const float*)d_in[11];
  const float* ln2b   = (const float*)d_in[12];
  const float* fc1w   = (const float*)d_in[13];
  const float* fc1b   = (const float*)d_in[14];
  const float* fc2w   = (const float*)d_in[15];
  const float* fc2b   = (const float*)d_in[16];

  char* ws = (char*)d_ws;
  size_t off = 0;
  auto carve = [&](size_t bytes) -> char* {
    char* p = ws + off;
    off = (off + bytes + 255) & ~(size_t)255;
    return p;
  };
  float*    xt   = (float*)   carve((size_t)B_ * N_TOK * C_ * 4);
  float*    xa   = (float*)   carve((size_t)B_ * MP * C_ * 4);
  _Float16* h16  = (_Float16*)carve((size_t)B_ * MP * C_ * 2);
  _Float16* q16  = (_Float16*)carve((size_t)B_ * NH * MP * HD * 2);
  _Float16* k16  = (_Float16*)carve((size_t)B_ * NH * MP * HD * 2);
  _Float16* vt16 = (_Float16*)carve((size_t)B_ * NH * MP * HD * 2);
  _Float16* o16  = (_Float16*)carve((size_t)B_ * MP * C_ * 2);
  float*    obuf = (float*)   carve((size_t)B_ * MP * C_ * 4);
  _Float16* m1   = (_Float16*)carve((size_t)B_ * MP * MLP * 2);
  float*    m2   = (float*)   carve((size_t)B_ * MP * C_ * 4);
  _Float16* wq16 = (_Float16*)carve((size_t)3 * C_ * C_ * 2);
  _Float16* wp16 = (_Float16*)carve((size_t)C_ * C_ * 2);
  _Float16* wf1  = (_Float16*)carve((size_t)MLP * C_ * 2);
  _Float16* wf2  = (_Float16*)carve((size_t)C_ * MLP * 2);
  int*      rank = (int*)     carve((size_t)B_ * N_TOK * 4);
  float*    sdrp = (float*)   carve((size_t)B_ * 4);
  float*    clsb = (float*)   carve((size_t)B_ * NKEEP * 4);

  float* out_cls = (float*)d_out;
  float* out_x   = out_cls + (size_t)B_ * C_;
  float* out_ga  = out_x + (size_t)B_ * C_ * N_TOK;

  // weight conversion
  k_f2h<<<(3 * C_ * C_ + 255) / 256, 256, 0, stream>>>(qkvw, wq16, 3 * C_ * C_);
  k_f2h<<<(C_ * C_ + 255) / 256, 256, 0, stream>>>(projw, wp16, C_ * C_);
  k_f2h<<<(MLP * C_ + 255) / 256, 256, 0, stream>>>(fc1w, wf1, MLP * C_);
  k_f2h<<<(C_ * MLP + 255) / 256, 256, 0, stream>>>(fc2w, wf2, C_ * MLP);

  // conv + tokens, ranking, prune/aggregate
  k_conv<<<(B_ * C_ * N_TOK + 255) / 256, 256, 0, stream>>>(x, conv_w, conv_b, xt);
  k_rank<<<(B_ * N_TOK + 255) / 256, 256, 0, stream>>>(ga, rank);
  k_sdrop<<<B_, 256, 0, stream>>>(ga, rank, sdrp);
  k_build_cls<<<(B_ * C_ + 255) / 256, 256, 0, stream>>>(cls, xa);
  k_build_xa<<<(B_ * N_TOK * C_ + 255) / 256, 256, 0, stream>>>(xt, rank, xa);
  k_rep<<<B_ * C_, 256, 0, stream>>>(xt, ga, rank, sdrp, xa);

  // transformer block
  k_ln<<<B_ * MP, 128, 0, stream>>>(xa, ln1g, ln1b, h16);
  { dim3 g(3 * C_ / 64, (B_ * MP) / 64);
    k_gemm_qkv<<<g, 128, 0, stream>>>(h16, wq16, q16, k16, vt16); }
  { dim3 g((QT_COUNT + 7) / 8, B_ * NH);
    k_attn<<<g, 256, 0, stream>>>(q16, k16, vt16, o16); }
  hipMemsetAsync(clsb, 0, (size_t)B_ * NKEEP * 4, stream);
  k_cls_attn<<<B_ * NH, 256, 0, stream>>>(q16, k16, clsb);
  { dim3 g(C_ / 64, (B_ * MP) / 64);
    k_gemm_bias<<<g, 128, 0, stream>>>(o16, wp16, projb, obuf, C_, C_); }
  k_add<<<(B_ * MP * C_ + 255) / 256, 256, 0, stream>>>(xa, obuf, (size_t)B_ * MP * C_);
  k_ln<<<B_ * MP, 128, 0, stream>>>(xa, ln2g, ln2b, h16);
  { dim3 g(MLP / 64, (B_ * MP) / 64);
    k_gemm_fc1<<<g, 128, 0, stream>>>(h16, wf1, fc1b, m1); }
  { dim3 g(C_ / 64, (B_ * MP) / 64);
    k_gemm_bias<<<g, 128, 0, stream>>>(m1, wf2, fc2b, m2, MLP, C_); }
  k_add<<<(B_ * MP * C_ + 255) / 256, 256, 0, stream>>>(xa, m2, (size_t)B_ * MP * C_);

  // outputs
  k_out_cls<<<(B_ * C_ + 255) / 256, 256, 0, stream>>>(xa, out_cls);
  k_out_patch<<<(B_ * C_ * N_TOK + 255) / 256, 256, 0, stream>>>(xa, xt, rank, obuf, m2, out_x);
  k_out_ga<<<(B_ * N_TOK + 255) / 256, 256, 0, stream>>>(ga, rank, clsb, out_ga);
}
